// MultiHeadSelfAttention_10917806866508
// MI455X (gfx1250) — compile-verified, matrix operand order checked by reading, not run on hardware
//
#include <hip/hip_runtime.h>

#define D_MODEL 1024
#define NHEAD 16
#define HEAD_DIM 64
#define BSZ 4
#define SEQ 2048
#define MTOT (BSZ * SEQ) /* 8192 */

typedef __attribute__((ext_vector_type(16))) __bf16 bf16x16;
typedef __attribute__((ext_vector_type(8)))  float  f32x8;
typedef __attribute__((ext_vector_type(4)))  float  f32x4;

// round-to-nearest-even f32 -> bf16
__device__ __forceinline__ __bf16 f2bf(float f) {
  unsigned u = __builtin_bit_cast(unsigned, f);
  u += 0x7FFFu + ((u >> 16) & 1u);
  unsigned short h = (unsigned short)(u >> 16);
  return __builtin_bit_cast(__bf16, h);
}

__device__ __forceinline__ f32x8 wmma_bf16(bf16x16 a, bf16x16 b, f32x8 c) {
  // (neg_a, A, neg_b, B, c_mod, C, reuse_a, reuse_b)
  return __builtin_amdgcn_wmma_f32_16x16x32_bf16(false, a, false, b, (short)0, c,
                                                 false, false);
}

// A/B fragment gather for V_WMMA_*_16X16X32 (wave32):
// lane L: row/col = L&15, half hh = L>>4; element e -> k = e + 8*hh + (e>=8 ? 8 : 0)
// Works for any row-major [16 x 32] view with leading dimension `ld`.
__device__ __forceinline__ bf16x16 frag_g_bf16(const __bf16* p, int ld, int row0,
                                               int k0, int lane) {
  int m = lane & 15, hh = lane >> 4;
  const __bf16* b0 = p + (size_t)(row0 + m) * ld + k0 + hh * 8;
  bf16x16 r;
#pragma unroll
  for (int e = 0; e < 8; ++e) r[e] = b0[e];
#pragma unroll
  for (int e = 0; e < 8; ++e) r[8 + e] = b0[16 + e];
  return r;
}

__device__ __forceinline__ bf16x16 frag_f32cvt(const float* p, int ld, int row0,
                                               int k0, int lane) {
  int m = lane & 15, hh = lane >> 4;
  const float* b0 = p + (size_t)(row0 + m) * ld + k0 + hh * 8;
  bf16x16 r;
#pragma unroll
  for (int e = 0; e < 8; ++e) r[e] = f2bf(b0[e]);
#pragma unroll
  for (int e = 0; e < 8; ++e) r[8 + e] = f2bf(b0[16 + e]);
  return r;
}

// ---------------------------------------------------------------------------
// Kernel 1: fused QKV projection. grid=(M/64, N/64, 3), block=128 (4 waves).
// y = x @ W^T + b ; Q,K stored bf16 [b,h,s,d]; V stored TRANSPOSED bf16 [b,h,d,s]
// ---------------------------------------------------------------------------
__global__ __launch_bounds__(128) void qkv_gemm_kernel(
    const float* __restrict__ x, const float* __restrict__ Wq,
    const float* __restrict__ bq, const float* __restrict__ Wk,
    const float* __restrict__ bk, const float* __restrict__ Wv,
    const float* __restrict__ bv, __bf16* __restrict__ qbuf,
    __bf16* __restrict__ kbuf, __bf16* __restrict__ vtbuf) {
  __shared__ float xs[64 * 32];
  __shared__ float ws[64 * 32];

  const int which = blockIdx.z;
  const float* W = which == 0 ? Wq : (which == 1 ? Wk : Wv);
  const float* bias = which == 0 ? bq : (which == 1 ? bk : bv);

  const int m0 = blockIdx.x * 64;
  const int n0 = blockIdx.y * 64;
  const int tid = threadIdx.x;
  const int lane = tid & 31;
  const int wv = tid >> 5;
  const int mo = (wv >> 1) * 32;  // wave's 32x32 quadrant
  const int no = (wv & 1) * 32;

  f32x8 acc[2][2] = {};

  for (int k0 = 0; k0 < D_MODEL; k0 += 32) {
    // cooperative 64x32 f32 tile loads (x rows, W rows) -> LDS (b128 path)
    const f32x4* xg = (const f32x4*)x;
    const f32x4* wg = (const f32x4*)W;
#pragma unroll
    for (int i = 0; i < 4; ++i) {
      int idx4 = tid + i * 128;  // 0..511 float4 slots (64 rows x 8)
      int r = idx4 >> 3;
      int c4 = idx4 & 7;
      ((f32x4*)xs)[idx4] = xg[((size_t)(m0 + r) * D_MODEL + k0) / 4 + c4];
      ((f32x4*)ws)[idx4] = wg[((size_t)(n0 + r) * D_MODEL + k0) / 4 + c4];
    }
    if (k0 + 32 < D_MODEL) {  // prefetch next slab (global_prefetch_b8)
      __builtin_prefetch(x + (size_t)(m0 + (tid >> 1)) * D_MODEL + k0 + 32 +
                             (tid & 1) * 16, 0, 1);
      __builtin_prefetch(W + (size_t)(n0 + (tid >> 1)) * D_MODEL + k0 + 32 +
                             (tid & 1) * 16, 0, 1);
    }
    __syncthreads();

    bf16x16 a0 = frag_f32cvt(xs, 32, mo + 0, 0, lane);
    bf16x16 a1 = frag_f32cvt(xs, 32, mo + 16, 0, lane);
    bf16x16 b0 = frag_f32cvt(ws, 32, no + 0, 0, lane);
    bf16x16 b1 = frag_f32cvt(ws, 32, no + 16, 0, lane);
    acc[0][0] = wmma_bf16(a0, b0, acc[0][0]);
    acc[0][1] = wmma_bf16(a0, b1, acc[0][1]);
    acc[1][0] = wmma_bf16(a1, b0, acc[1][0]);
    acc[1][1] = wmma_bf16(a1, b1, acc[1][1]);
    __syncthreads();
  }

  // epilogue: bias + bf16 store in per-head layouts
  const int hh = lane >> 4, nl = lane & 15;
#pragma unroll
  for (int ti = 0; ti < 2; ++ti)
#pragma unroll
    for (int tj = 0; tj < 2; ++tj)
#pragma unroll
      for (int r = 0; r < 8; ++r) {
        int gm = m0 + mo + ti * 16 + r + 8 * hh;  // D layout: M = r + 8*(L/16)
        int gn = n0 + no + tj * 16 + nl;          // N = L%16
        float v = acc[ti][tj][r] + bias[gn];
        int b = gm >> 11, s = gm & 2047;
        int h = gn >> 6, d = gn & 63;
        __bf16 o = f2bf(v);
        if (which == 0)
          qbuf[(((size_t)(b * NHEAD + h)) * SEQ + s) * HEAD_DIM + d] = o;
        else if (which == 1)
          kbuf[(((size_t)(b * NHEAD + h)) * SEQ + s) * HEAD_DIM + d] = o;
        else
          vtbuf[(((size_t)(b * NHEAD + h)) * HEAD_DIM + d) * SEQ + s] = o;
      }
}

// ---------------------------------------------------------------------------
// Kernel 2: causal flash attention. grid=(S/64, B*H), block=128 (4 waves).
// Each wave owns 16 query rows; chunk loop over 32 kv positions is wave-uniform
// (no cross-wave barriers). LDS is wave-private (DS ops are in-order per wave).
// ---------------------------------------------------------------------------
__global__ __launch_bounds__(128) void attn_kernel(
    const __bf16* __restrict__ qbuf, const __bf16* __restrict__ kbuf,
    const __bf16* __restrict__ vtbuf, __bf16* __restrict__ ctxbuf) {
  __shared__ float sS[4][16][32];
  __shared__ __bf16 sP[4][16][32];
  __shared__ float sFac[4][16];
  __shared__ float sL[4][16];

  const int tid = threadIdx.x;
  const int lane = tid & 31;
  const int wv = tid >> 5;
  const int bh = blockIdx.y;
  const int b = bh >> 4, h = bh & 15;
  const int qbase = blockIdx.x * 64 + wv * 16;

  const __bf16* Q = qbuf + (size_t)bh * SEQ * HEAD_DIM;
  const __bf16* K = kbuf + (size_t)bh * SEQ * HEAD_DIM;
  const __bf16* Vt = vtbuf + (size_t)bh * HEAD_DIM * SEQ;

  bf16x16 aq0 = frag_g_bf16(Q, HEAD_DIM, qbase, 0, lane);
  bf16x16 aq1 = frag_g_bf16(Q, HEAD_DIM, qbase, 32, lane);

  f32x8 acc[4] = {};
  float m_i = -__builtin_inff();  // valid in lanes 0..15 (lane == row)
  float l_i = 0.f;

  const int nchunk = (qbase + 15) / 32 + 1;  // t0 <= qbase always holds
  for (int c = 0; c < nchunk; ++c) {
    const int t0 = c * 32;
    const int hh = lane >> 4, nl = lane & 15;

    // S = (Q K^T) * 1/sqrt(64) for this 16x32 chunk
#pragma unroll
    for (int j = 0; j < 2; ++j) {
      bf16x16 bk0 = frag_g_bf16(K, HEAD_DIM, t0 + 16 * j, 0, lane);
      bf16x16 bk1 = frag_g_bf16(K, HEAD_DIM, t0 + 16 * j, 32, lane);
      f32x8 s = {};
      s = wmma_bf16(aq0, bk0, s);
      s = wmma_bf16(aq1, bk1, s);
#pragma unroll
      for (int r = 0; r < 8; ++r)
        sS[wv][r + 8 * hh][16 * j + nl] = s[r] * 0.125f;
    }

    // online softmax: lane < 16 owns one row
    if (lane < 16) {
      const int row = lane;
      const int qrow = qbase + row;
      float cm = -__builtin_inff();
#pragma unroll
      for (int j = 0; j < 32; ++j) {
        float v = sS[wv][row][j];
        if (t0 + j > qrow) v = -__builtin_inff();
        sS[wv][row][j] = v;
        cm = fmaxf(cm, v);
      }
      float newm = fmaxf(m_i, cm);
      float fac = __expf(m_i - newm);
      float sum = 0.f;
#pragma unroll
      for (int j = 0; j < 32; ++j) {
        float p = __expf(sS[wv][row][j] - newm);
        sum += p;
        sP[wv][row][j] = f2bf(p);
      }
      l_i = l_i * fac + sum;
      m_i = newm;
      sFac[wv][row] = fac;
      sL[wv][row] = l_i;
    }

    // rescale accumulators by per-row factor, then acc += P @ V
    float fr[8];
#pragma unroll
    for (int r = 0; r < 8; ++r) fr[r] = sFac[wv][r + 8 * hh];
#pragma unroll
    for (int t = 0; t < 4; ++t)
#pragma unroll
      for (int r = 0; r < 8; ++r) acc[t][r] *= fr[r];

    bf16x16 pfrag = frag_g_bf16(&sP[wv][0][0], 32, 0, 0, lane);
#pragma unroll
    for (int t = 0; t < 4; ++t) {
      bf16x16 bv = frag_g_bf16(Vt, SEQ, t * 16, t0, lane);  // Vt rows = d
      acc[t] = wmma_bf16(pfrag, bv, acc[t]);
    }
  }

  // normalize and store ctx bf16 as [b, s, h*64+d]
  {
    const int hh = lane >> 4, nl = lane & 15;
    float linv[8];
#pragma unroll
    for (int r = 0; r < 8; ++r) linv[r] = 1.0f / sL[wv][r + 8 * hh];
#pragma unroll
    for (int t = 0; t < 4; ++t)
#pragma unroll
      for (int r = 0; r < 8; ++r) {
        int s = qbase + r + 8 * hh;
        int d = t * 16 + nl;
        ctxbuf[((size_t)(b * SEQ + s)) * D_MODEL + h * HEAD_DIM + d] =
            f2bf(acc[t][r] * linv[r]);
      }
  }
}

// ---------------------------------------------------------------------------
// Kernel 3: output projection out = ctx @ Wo^T + bo (f32 out).
// ---------------------------------------------------------------------------
__global__ __launch_bounds__(128) void oproj_kernel(
    const __bf16* __restrict__ ctxbuf, const float* __restrict__ Wo,
    const float* __restrict__ bo, float* __restrict__ out) {
  const int m0 = blockIdx.x * 64;
  const int n0 = blockIdx.y * 64;
  const int tid = threadIdx.x;
  const int lane = tid & 31;
  const int wv = tid >> 5;
  const int mo = (wv >> 1) * 32, no = (wv & 1) * 32;

  f32x8 acc[2][2] = {};
  for (int k0 = 0; k0 < D_MODEL; k0 += 32) {
    bf16x16 a0 = frag_g_bf16(ctxbuf, D_MODEL, m0 + mo, k0, lane);
    bf16x16 a1 = frag_g_bf16(ctxbuf, D_MODEL, m0 + mo + 16, k0, lane);
    bf16x16 b0 = frag_f32cvt(Wo, D_MODEL, n0 + no, k0, lane);
    bf16x16 b1 = frag_f32cvt(Wo, D_MODEL, n0 + no + 16, k0, lane);
    acc[0][0] = wmma_bf16(a0, b0, acc[0][0]);
    acc[0][1] = wmma_bf16(a0, b1, acc[0][1]);
    acc[1][0] = wmma_bf16(a1, b0, acc[1][0]);
    acc[1][1] = wmma_bf16(a1, b1, acc[1][1]);
  }
  const int hh = lane >> 4, nl = lane & 15;
#pragma unroll
  for (int ti = 0; ti < 2; ++ti)
#pragma unroll
    for (int tj = 0; tj < 2; ++tj)
#pragma unroll
      for (int r = 0; r < 8; ++r) {
        int gm = m0 + mo + ti * 16 + r + 8 * hh;
        int gn = n0 + no + tj * 16 + nl;
        out[(size_t)gm * D_MODEL + gn] = acc[ti][tj][r] + bo[gn];
      }
}

extern "C" void kernel_launch(void* const* d_in, const int* in_sizes, int n_in,
                              void* d_out, int out_size, void* d_ws,
                              size_t ws_size, hipStream_t stream) {
  const float* x = (const float*)d_in[0];
  // d_in[1] = causal mask (implicit in kernel)
  const float* Wq = (const float*)d_in[2];
  const float* bq = (const float*)d_in[3];
  const float* Wk = (const float*)d_in[4];
  const float* bk = (const float*)d_in[5];
  const float* Wv = (const float*)d_in[6];
  const float* bv = (const float*)d_in[7];
  const float* Wo = (const float*)d_in[8];
  const float* bo = (const float*)d_in[9];
  float* out = (float*)d_out;

  const size_t elems = (size_t)BSZ * SEQ * D_MODEL;  // 8M
  __bf16* qbuf = (__bf16*)d_ws;                      // 16 MB each
  __bf16* kbuf = qbuf + elems;
  __bf16* vtbuf = kbuf + elems;
  __bf16* ctxb = vtbuf + elems;  // total 64 MB scratch

  qkv_gemm_kernel<<<dim3(MTOT / 64, D_MODEL / 64, 3), 128, 0, stream>>>(
      x, Wq, bq, Wk, bk, Wv, bv, qbuf, kbuf, vtbuf);
  attn_kernel<<<dim3(SEQ / 64, BSZ * NHEAD), 128, 0, stream>>>(qbuf, kbuf,
                                                               vtbuf, ctxb);
  oproj_kernel<<<dim3(MTOT / 64, D_MODEL / 64), 128, 0, stream>>>(ctxb, Wo, bo,
                                                                  out);
}
